// MultiheadAttention_vanilla_91122026152534
// MI455X (gfx1250) — matrix operand order repaired and checked by reading, not compiled, not run on hardware
//
#include <hip/hip_runtime.h>
#include <hip/hip_bf16.h>

typedef __attribute__((ext_vector_type(16))) __bf16 v16bf;
typedef __attribute__((ext_vector_type(8)))  float  v8f;

#define EMBED 1024
#define NH    16
#define HD    64
#define BATCH 2
#define SEQ   2048
#define MROWS (BATCH * SEQ)

union FragB16 { v16bf v; uint4 q[2]; unsigned u[8]; };

__device__ __forceinline__ unsigned short f2bf(float f) {
    unsigned u = __builtin_bit_cast(unsigned, f);
    u += 0x7FFFu + ((u >> 16) & 1u);          // round-to-nearest-even
    return (unsigned short)(u >> 16);
}
__device__ __forceinline__ unsigned packbf(float lo, float hi) {
    return (unsigned)f2bf(lo) | ((unsigned)f2bf(hi) << 16);
}

// ---------------------------------------------------------------------------
// One-time f32 -> bf16 conversion (4 elements / thread)
// ---------------------------------------------------------------------------
__global__ __launch_bounds__(256)
void cvt_bf16_kernel(const float* __restrict__ in,
                     unsigned short* __restrict__ out, int n4) {
    int i = blockIdx.x * 256 + threadIdx.x;
    if (i < n4) {
        float4 f = ((const float4*)in)[i];
        uint2 p;
        p.x = packbf(f.x, f.y);
        p.y = packbf(f.z, f.w);
        ((uint2*)out)[i] = p;
    }
}

// ---------------------------------------------------------------------------
// bf16 GEMM: out = A @ W^T + bias.  A:[4096,K] bf16, W:[1024,K] bf16,
// 32x64 tile per wave (8 WMMAs / 12 b128 loads per K-step).
// mode 0: bf16 [B,H,N,D] (Q,K)  mode 1: bf16 [B,H,D,N] (V^T)  mode 2: f32 [M,C]
// ---------------------------------------------------------------------------
__global__ __launch_bounds__(256)
void gemm_bf16_kernel(const unsigned short* __restrict__ A,
                      const unsigned short* __restrict__ W,
                      const float* __restrict__ bias,
                      void* __restrict__ out, int mode) {
    const int lane = threadIdx.x & 31;
    const int wid  = threadIdx.x >> 5;
    const int gx = blockIdx.x & 3;
    const int gy = blockIdx.x >> 2;
    const int r  = lane & 15;
    const int h  = lane >> 4;
    const int row0 = gy * 64 + (wid >> 2) * 32;
    const int col0 = gx * 256 + (wid & 3) * 64;

    const unsigned short* arow0 = A + (size_t)(row0 + r) * EMBED;
    const unsigned short* arow1 = arow0 + 16 * EMBED;
    const unsigned short* wrow0 = W + (size_t)(col0 + r) * EMBED;
    const unsigned short* wrow1 = wrow0 + 16 * EMBED;
    const unsigned short* wrow2 = wrow0 + 32 * EMBED;
    const unsigned short* wrow3 = wrow0 + 48 * EMBED;

    v8f acc[2][4];
#pragma unroll
    for (int i = 0; i < 2; ++i)
#pragma unroll
        for (int j = 0; j < 4; ++j) acc[i][j] = (v8f){};

    for (int kk = 0; kk < EMBED; kk += 32) {
        FragB16 a0, a1, b0, b1, b2, b3;
        a0.q[0] = *(const uint4*)(arow0 + kk + 16 * h);
        a0.q[1] = *(const uint4*)(arow0 + kk + 16 * h + 8);
        a1.q[0] = *(const uint4*)(arow1 + kk + 16 * h);
        a1.q[1] = *(const uint4*)(arow1 + kk + 16 * h + 8);
        b0.q[0] = *(const uint4*)(wrow0 + kk + 16 * h);
        b0.q[1] = *(const uint4*)(wrow0 + kk + 16 * h + 8);
        b1.q[0] = *(const uint4*)(wrow1 + kk + 16 * h);
        b1.q[1] = *(const uint4*)(wrow1 + kk + 16 * h + 8);
        b2.q[0] = *(const uint4*)(wrow2 + kk + 16 * h);
        b2.q[1] = *(const uint4*)(wrow2 + kk + 16 * h + 8);
        b3.q[0] = *(const uint4*)(wrow3 + kk + 16 * h);
        b3.q[1] = *(const uint4*)(wrow3 + kk + 16 * h + 8);

        acc[0][0] = __builtin_amdgcn_wmma_f32_16x16x32_bf16(false, a0.v, false, b0.v, (short)0, acc[0][0], false, false);
        acc[0][1] = __builtin_amdgcn_wmma_f32_16x16x32_bf16(false, a0.v, false, b1.v, (short)0, acc[0][1], false, false);
        acc[0][2] = __builtin_amdgcn_wmma_f32_16x16x32_bf16(false, a0.v, false, b2.v, (short)0, acc[0][2], false, false);
        acc[0][3] = __builtin_amdgcn_wmma_f32_16x16x32_bf16(false, a0.v, false, b3.v, (short)0, acc[0][3], false, false);
        acc[1][0] = __builtin_amdgcn_wmma_f32_16x16x32_bf16(false, a1.v, false, b0.v, (short)0, acc[1][0], false, false);
        acc[1][1] = __builtin_amdgcn_wmma_f32_16x16x32_bf16(false, a1.v, false, b1.v, (short)0, acc[1][1], false, false);
        acc[1][2] = __builtin_amdgcn_wmma_f32_16x16x32_bf16(false, a1.v, false, b2.v, (short)0, acc[1][2], false, false);
        acc[1][3] = __builtin_amdgcn_wmma_f32_16x16x32_bf16(false, a1.v, false, b3.v, (short)0, acc[1][3], false, false);
    }

#pragma unroll
    for (int ct = 0; ct < 4; ++ct) {
        const int col = col0 + ct * 16 + r;
        const float bv = bias[col];
        const int hh = col >> 6;
        const int d  = col & (HD - 1);
#pragma unroll
        for (int rt = 0; rt < 2; ++rt) {
#pragma unroll
            for (int rr = 0; rr < 8; ++rr) {
                int m = row0 + rt * 16 + rr + 8 * h;
                float val = acc[rt][ct][rr] + bv;
                if (mode == 2) {
                    ((float*)out)[(size_t)m * EMBED + col] = val;
                } else {
                    int b = m >> 11;
                    int n = m & (SEQ - 1);
                    size_t off = (mode == 1)
                        ? ((size_t)((b * NH + hh) * HD + d)) * SEQ + n
                        : ((size_t)((b * NH + hh) * SEQ + n)) * HD + d;
                    ((unsigned short*)out)[off] = f2bf(val);
                }
            }
        }
    }
}

// ---------------------------------------------------------------------------
// Flash attention. Block = 4 waves sharing one (b,h); each wave owns a
// 16-query tile. 32-key K/V chunks are staged into LDS once per block via
// async global->LDS copies (ASYNCcnt), then consumed as WMMA fragments
// directly from LDS. Q bf16 [B,H,N,D]; K bf16 [B,H,N,D]; V bf16 [B,H,D,N];
// O bf16 [B,N,C].
// ---------------------------------------------------------------------------
__global__ __launch_bounds__(128)
void attn_kernel(const unsigned short* __restrict__ Q,
                 const unsigned short* __restrict__ K,
                 const unsigned short* __restrict__ Vt,
                 unsigned short* __restrict__ O) {
    __shared__ __align__(16) unsigned short kbuf[32 * 64];   // [key][d]
    __shared__ __align__(16) unsigned short vbuf[64 * 32];   // [d][key]
    __shared__ __align__(16) unsigned short pbuf[4][16 * 32];

    const int tid  = threadIdx.x;
    const int wid  = tid >> 5;
    const int lane = tid & 31;
    const int hb = blockIdx.x >> 5;              // b*NH + h
    const int qt = (blockIdx.x & 31) * 4 + wid;  // query tile of 16
    const int b  = hb >> 4;
    const int h_ = hb & 15;
    const int r  = lane & 15;
    const int hf = lane >> 4;
    unsigned short* pl = pbuf[wid];

    // LDS byte offsets for async copies (flat shared addr low bits = LDS offset)
    const unsigned klds = (unsigned)(uintptr_t)kbuf + tid * 16;
    const int vd = tid >> 2, vp = tid & 3;
    const unsigned vlds = (unsigned)(uintptr_t)vbuf + (vd * 32 + vp * 8) * 2;

    const unsigned short* kbase = K + (size_t)hb * SEQ * HD;          // + j0*HD
    const unsigned short* vbase = Vt + ((size_t)hb * HD + vd) * SEQ;  // + j0

    // Q A-fragments for d=0..31 and d=32..63 (kept resident)
    const unsigned short* qrow = Q + ((size_t)hb * SEQ + qt * 16 + r) * HD;
    FragB16 qlo, qhi;
    qlo.q[0] = *(const uint4*)(qrow + 16 * hf);
    qlo.q[1] = *(const uint4*)(qrow + 16 * hf + 8);
    qhi.q[0] = *(const uint4*)(qrow + 32 + 16 * hf);
    qhi.q[1] = *(const uint4*)(qrow + 32 + 16 * hf + 8);

    float mrow[8], lrow[8];
    v8f o0 = {}, o1 = {}, o2 = {}, o3 = {};
#pragma unroll
    for (int i = 0; i < 8; ++i) { mrow[i] = -1e30f; lrow[i] = 0.0f; }

    for (int j0 = 0; j0 < SEQ; j0 += 32) {
        // --- async stage of 32-key K (4KB) and V (4KB) chunks into LDS ---
        {
            const unsigned short* kg = kbase + j0 * HD + tid * 8;
            const unsigned short* vg = vbase + j0 + vp * 8;
            asm volatile("global_load_async_to_lds_b128 %0, %1, off"
                         :: "v"(klds), "v"(kg) : "memory");
            asm volatile("global_load_async_to_lds_b128 %0, %1, off"
                         :: "v"(klds + 2048u), "v"(kg + 1024) : "memory");
            asm volatile("global_load_async_to_lds_b128 %0, %1, off"
                         :: "v"(vlds), "v"(vg) : "memory");
            asm volatile("global_load_async_to_lds_b128 %0, %1, off"
                         :: "v"(vlds + 2048u), "v"(vg + 32 * SEQ) : "memory");
            asm volatile("s_wait_asynccnt 0" ::: "memory");
        }
        __syncthreads();

        // --- S = (Q K^T) * 1/sqrt(D), two 16x16 key tiles, K from LDS ---
        FragB16 k0lo, k0hi, k1lo, k1hi;
        const unsigned short* krow0 = kbuf + r * HD;          // key j0+r
        const unsigned short* krow1 = kbuf + (16 + r) * HD;   // key j0+16+r
        k0lo.q[0] = *(const uint4*)(krow0 + 16 * hf);
        k0lo.q[1] = *(const uint4*)(krow0 + 16 * hf + 8);
        k0hi.q[0] = *(const uint4*)(krow0 + 32 + 16 * hf);
        k0hi.q[1] = *(const uint4*)(krow0 + 32 + 16 * hf + 8);
        k1lo.q[0] = *(const uint4*)(krow1 + 16 * hf);
        k1lo.q[1] = *(const uint4*)(krow1 + 16 * hf + 8);
        k1hi.q[0] = *(const uint4*)(krow1 + 32 + 16 * hf);
        k1hi.q[1] = *(const uint4*)(krow1 + 32 + 16 * hf + 8);

        v8f s0 = {}, s1 = {};
        s0 = __builtin_amdgcn_wmma_f32_16x16x32_bf16(false, qlo.v, false, k0lo.v, (short)0, s0, false, false);
        s0 = __builtin_amdgcn_wmma_f32_16x16x32_bf16(false, qhi.v, false, k0hi.v, (short)0, s0, false, false);
        s1 = __builtin_amdgcn_wmma_f32_16x16x32_bf16(false, qlo.v, false, k1lo.v, (short)0, s1, false, false);
        s1 = __builtin_amdgcn_wmma_f32_16x16x32_bf16(false, qhi.v, false, k1hi.v, (short)0, s1, false, false);

        // --- online softmax: rows live across 16-lane groups ---
        float mnew[8], corr[8];
#pragma unroll
        for (int rr = 0; rr < 8; ++rr) {
            float a = s0[rr] * 0.125f;
            float c = s1[rr] * 0.125f;
            s0[rr] = a; s1[rr] = c;
            float mx = fmaxf(a, c);
            mx = fmaxf(mx, __shfl_xor(mx, 1));
            mx = fmaxf(mx, __shfl_xor(mx, 2));
            mx = fmaxf(mx, __shfl_xor(mx, 4));
            mx = fmaxf(mx, __shfl_xor(mx, 8));
            mnew[rr] = fmaxf(mrow[rr], mx);
            corr[rr] = __expf(mrow[rr] - mnew[rr]);
            mrow[rr] = mnew[rr];
        }
#pragma unroll
        for (int rr = 0; rr < 8; ++rr) {
            float p0 = __expf(s0[rr] - mnew[rr]);
            float p1 = __expf(s1[rr] - mnew[rr]);
            float rs = p0 + p1;
            rs += __shfl_xor(rs, 1);
            rs += __shfl_xor(rs, 2);
            rs += __shfl_xor(rs, 4);
            rs += __shfl_xor(rs, 8);
            lrow[rr] = lrow[rr] * corr[rr] + rs;
            int prow = rr + 8 * hf;
            pl[prow * 32 + r]      = f2bf(p0);   // keys j0..j0+15
            pl[prow * 32 + 16 + r] = f2bf(p1);   // keys j0+16..j0+31
            o0[rr] *= corr[rr];
            o1[rr] *= corr[rr];
            o2[rr] *= corr[rr];
            o3[rr] *= corr[rr];
        }
        asm volatile("s_wait_dscnt 0" ::: "memory");

        // P (16x32) back as A-fragment (transpose through LDS)
        FragB16 pf;
        pf.q[0] = *(const uint4*)(pl + r * 32 + 16 * hf);
        pf.q[1] = *(const uint4*)(pl + r * 32 + 16 * hf + 8);

        // --- O += P @ V, 4 d-tiles of 16, V from LDS ---
#pragma unroll
        for (int t = 0; t < 4; ++t) {
            FragB16 fv;
            const unsigned short* vrow = vbuf + (t * 16 + r) * 32;
            fv.q[0] = *(const uint4*)(vrow + 16 * hf);
            fv.q[1] = *(const uint4*)(vrow + 16 * hf + 8);
            v8f& oo = (t == 0) ? o0 : (t == 1) ? o1 : (t == 2) ? o2 : o3;
            oo = __builtin_amdgcn_wmma_f32_16x16x32_bf16(
                false, pf.v, false, fv.v, (short)0, oo, false, false);
        }
        __syncthreads();   // protect K/V LDS chunk before next stage
    }

    // epilogue: normalize and store bf16 into [B,N,C]
#pragma unroll
    for (int rr = 0; rr < 8; ++rr) {
        float inv = 1.0f / lrow[rr];
        int qg = qt * 16 + rr + 8 * hf;
        unsigned short* orow = O + ((size_t)(b * SEQ + qg)) * EMBED + h_ * HD;
        orow[0 * 16 + r] = f2bf(o0[rr] * inv);
        orow[1 * 16 + r] = f2bf(o1[rr] * inv);
        orow[2 * 16 + r] = f2bf(o2[rr] * inv);
        orow[3 * 16 + r] = f2bf(o3[rr] * inv);
    }
}

// ---------------------------------------------------------------------------
extern "C" void kernel_launch(void* const* d_in, const int* in_sizes, int n_in,
                              void* d_out, int out_size, void* d_ws, size_t ws_size,
                              hipStream_t stream) {
    (void)in_sizes; (void)n_in; (void)out_size; (void)ws_size;
    const float* query = (const float*)d_in[0];
    const float* key   = (const float*)d_in[1];
    const float* value = (const float*)d_in[2];
    const float* Wq = (const float*)d_in[3];
    const float* bq = (const float*)d_in[4];
    const float* Wk = (const float*)d_in[5];
    const float* bk = (const float*)d_in[6];
    const float* Wv = (const float*)d_in[7];
    const float* bv = (const float*)d_in[8];
    const float* Wo = (const float*)d_in[9];
    const float* bo = (const float*)d_in[10];

    const size_t act  = (size_t)MROWS * EMBED;
    const size_t wele = (size_t)EMBED * EMBED;
    unsigned short* xbuf = (unsigned short*)d_ws;
    unsigned short* wbuf = xbuf + act;
    unsigned short* qb   = wbuf + wele;
    unsigned short* kbuf = qb + act;
    unsigned short* vt   = kbuf + act;
    unsigned short* ob   = vt + act;

    const int actBlocks = (int)(act / 4 / 256);
    const int wBlocks   = (int)(wele / 4 / 256);
    const int gemmGrid  = (MROWS / 64) * (EMBED / 256);

    cvt_bf16_kernel<<<actBlocks, 256, 0, stream>>>(query, xbuf, (int)(act / 4));
    cvt_bf16_kernel<<<wBlocks,   256, 0, stream>>>(Wq, wbuf, (int)(wele / 4));
    gemm_bf16_kernel<<<gemmGrid, 256, 0, stream>>>(xbuf, wbuf, bq, qb, 0);

    cvt_bf16_kernel<<<actBlocks, 256, 0, stream>>>(key, xbuf, (int)(act / 4));
    cvt_bf16_kernel<<<wBlocks,   256, 0, stream>>>(Wk, wbuf, (int)(wele / 4));
    gemm_bf16_kernel<<<gemmGrid, 256, 0, stream>>>(xbuf, wbuf, bk, kbuf, 0);

    cvt_bf16_kernel<<<actBlocks, 256, 0, stream>>>(value, xbuf, (int)(act / 4));
    cvt_bf16_kernel<<<wBlocks,   256, 0, stream>>>(Wv, wbuf, (int)(wele / 4));
    gemm_bf16_kernel<<<gemmGrid, 256, 0, stream>>>(xbuf, wbuf, bv, vt, 1);

    attn_kernel<<<1024, 128, 0, stream>>>(qb, kbuf, vt, ob);

    cvt_bf16_kernel<<<wBlocks, 256, 0, stream>>>(Wo, wbuf, (int)(wele / 4));
    gemm_bf16_kernel<<<gemmGrid, 256, 0, stream>>>(ob, wbuf, bo, d_out, 2);
}
